// ConfigurableLSTMModel_26594437497060
// MI455X (gfx1250) — matrix operand order repaired and checked by reading, not compiled, not run on hardware
//
#include <hip/hip_runtime.h>
#include <hip/hip_bf16.h>
#include <math.h>

// ---------------- problem constants ----------------
constexpr int CB   = 256;   // batch
constexpr int CD   = 16;    // feature dim
constexpr int TIN  = 512;   // encoder steps
constexpr int TOUT = 200;   // decoder steps
constexpr int CH   = 512;   // hidden
constexpr int G4   = 4 * CH;      // 2048 gate rows (i,f,g,o)
constexpr int KCE  = CH + 32;     // 544 = 512 (h) + 16 (x) + 16 zero-pad (K mult of 32)
constexpr int NWG  = 128;         // persistent grid (8 batch-groups x 16 unit-strips)
constexpr int BT   = 32;          // batch tile per WG
constexpr int UT   = 32;          // hidden-unit tile per WG

// ---------------- workspace layout (bytes) ----------------
constexpr size_t OFF_MEAN = 0;                                  // 4096 f32
constexpr size_t OFF_STD  = OFF_MEAN + 4096 * 4;                // 4096 f32
constexpr size_t OFF_XN   = OFF_STD  + 4096 * 4;                // TIN*CB*CD bf16
constexpr size_t OFF_WE   = OFF_XN   + (size_t)TIN * CB * CD * 2;   // G4*KCE bf16
constexpr size_t OFF_WD   = OFF_WE   + (size_t)G4 * KCE * 2;        // G4*CH bf16
constexpr size_t OFF_WP   = OFF_WD   + (size_t)G4 * CH * 2;         // CD*CH bf16
constexpr size_t OFF_BE   = OFF_WP   + (size_t)CD * CH * 2;         // G4 f32
constexpr size_t OFF_BD   = OFF_BE   + (size_t)G4 * 4;              // G4 f32
constexpr size_t OFF_H    = OFF_BD   + (size_t)G4 * 4;              // 2*CB*CH bf16 (double buffer)
constexpr size_t OFF_SYNC = OFF_H    + (size_t)2 * CB * CH * 2;     // 2 u32

// ---------------- types ----------------
typedef __bf16 bf16;
typedef __attribute__((ext_vector_type(16))) __bf16 v16bf;
typedef __attribute__((ext_vector_type(8)))  float  v8f;
typedef __attribute__((ext_vector_type(4)))  unsigned v4u;

union Ubf { v4u u[2]; v16bf v; };

__device__ inline v8f wmma_bf16(v16bf a, v16bf b, v8f c) {
  // D = A(16x32 bf16) x B(32x16 bf16) + C(16x16 f32)
  return __builtin_amdgcn_wmma_f32_16x16x32_bf16(false, a, false, b, (short)0, c,
                                                 false, false);
}

// One 16-byte chunk of the A matrix row: K<512 -> h, 512<=K<528 -> x_t, else 0
__device__ inline v4u ldchunk(const bf16* __restrict__ h,
                              const bf16* __restrict__ xrow, int m, int k) {
  if (k < CH)       return *(const v4u*)(h + (size_t)m * CH + k);
  if (k < CH + CD)  return *(const v4u*)(xrow + (size_t)m * CD + (k - CH));
  v4u z = {0u, 0u, 0u, 0u};
  return z;
}

// A fragment (16x32 bf16), ISA layout: lanes 0-15 rows M, VGPR0-3 K=koff..koff+7,
// VGPR4-7 K=16+koff..; koff = 8 for lanes 16-31.
__device__ inline v16bf load_a(const bf16* __restrict__ h,
                               const bf16* __restrict__ xrow,
                               int B0, int mt, int kb, int lane) {
  int m    = B0 + mt * 16 + (lane & 15);
  int koff = (lane & 16) ? 8 : 0;
  Ubf u;
  u.u[0] = ldchunk(h, xrow, m, kb + koff);
  u.u[1] = ldchunk(h, xrow, m, kb + 16 + koff);
  return u.v;
}

// B fragment (32x16 bf16) from row-major (N,K) weights: lane holds column N=lane&15,
// K=koff..koff+7 and K=16+koff..16+koff+7 (koff = 8 for lanes 16-31) -> two b128 loads.
__device__ inline v16bf load_b16(const bf16* __restrict__ wrow, int kb, int lane) {
  int kbase = kb + ((lane & 16) ? 8 : 0);
  Ubf u;
  u.u[0] = *(const v4u*)(wrow + kbase);
  u.u[1] = *(const v4u*)(wrow + kbase + 16);
  return u.v;
}

__device__ inline float sigmoidf_(float x) { return 1.0f / (1.0f + __expf(-x)); }

// ---------------- device-wide barrier (persistent grid, all WGs resident) ----
__device__ inline void grid_sync(unsigned* cnt, unsigned* gen) {
  __threadfence();
  __syncthreads();
  if (threadIdx.x == 0) {
    unsigned g = __hip_atomic_load(gen, __ATOMIC_RELAXED, __HIP_MEMORY_SCOPE_AGENT);
    unsigned a = __hip_atomic_fetch_add(cnt, 1u, __ATOMIC_ACQ_REL, __HIP_MEMORY_SCOPE_AGENT);
    if (a == NWG - 1) {
      __hip_atomic_store(cnt, 0u, __ATOMIC_RELAXED, __HIP_MEMORY_SCOPE_AGENT);
      __hip_atomic_fetch_add(gen, 1u, __ATOMIC_ACQ_REL, __HIP_MEMORY_SCOPE_AGENT);
    } else {
      while (__hip_atomic_load(gen, __ATOMIC_ACQUIRE, __HIP_MEMORY_SCOPE_AGENT) == g) {
        __builtin_amdgcn_s_sleep(2);
      }
    }
  }
  __syncthreads();
  __threadfence();
}

// ================= prep kernels =================

// per-(b,d) mean / stdev over time (one wave per row)
__global__ void stats_kernel(const float* __restrict__ x, float* __restrict__ mean,
                             float* __restrict__ stdev) {
  int row  = blockIdx.x * 4 + (threadIdx.x >> 5);  // 0..4095
  int lane = threadIdx.x & 31;
  const float* p = x + (size_t)row * TIN;
  float s = 0.f, s2 = 0.f;
  for (int t = lane; t < TIN; t += 32) { float v = p[t]; s += v; s2 += v * v; }
  for (int off = 16; off > 0; off >>= 1) {
    s  += __shfl_down(s,  off, 32);
    s2 += __shfl_down(s2, off, 32);
  }
  if (lane == 0) {
    float m   = s / TIN;
    float var = s2 / TIN - m * m;
    mean[row]  = m;
    stdev[row] = sqrtf(var + 1e-5f);
  }
}

// normalized input, stored (t, b, d) bf16
__global__ void xn_kernel(const float* __restrict__ x, const float* __restrict__ mean,
                          const float* __restrict__ stdev, const float* __restrict__ rw,
                          const float* __restrict__ rb, bf16* __restrict__ xn) {
  int e = blockIdx.x * blockDim.x + threadIdx.x;
  if (e >= TIN * CB * CD) return;
  int d = e & 15, b = (e >> 4) & 255, t = e >> 12;
  float v = x[(size_t)b * CD * TIN + (size_t)d * TIN + t];
  int r = b * CD + d;
  xn[e] = (bf16)((v - mean[r]) / stdev[r] * rw[d] + rb[d]);
}

// encoder weights: row r (gate row), cols [0,512)=W_hh, [512,528)=W_ih, [528,544)=0
__global__ void pack_enc_kernel(const float* __restrict__ Whh, const float* __restrict__ Wih,
                                bf16* __restrict__ We) {
  int e = blockIdx.x * blockDim.x + threadIdx.x;
  if (e >= G4 * KCE) return;
  int r = e / KCE, k = e % KCE;
  float v = 0.f;
  if (k < CH)            v = Whh[(size_t)r * CH + k];
  else if (k < CH + CD)  v = Wih[(size_t)r * CD + (k - CH)];
  We[e] = (bf16)v;
}

// decoder weights: W_dec = W_hh + W_ih @ W_proj  (feedback folded into recurrence)
__global__ void pack_dec_kernel(const float* __restrict__ Whh, const float* __restrict__ Wih,
                                const float* __restrict__ Wp, bf16* __restrict__ Wd) {
  int e = blockIdx.x * blockDim.x + threadIdx.x;
  if (e >= G4 * CH) return;
  int r = e / CH, k = e % CH;
  float acc = Whh[e];
  #pragma unroll
  for (int j = 0; j < CD; ++j) acc += Wih[(size_t)r * CD + j] * Wp[(size_t)j * CH + k];
  Wd[e] = (bf16)acc;
}

// W_proj->bf16, biases (b_enc = b_ih+b_hh, b_dec = b_enc + W_ih@b_proj), sync init
__global__ void pack_misc_kernel(const float* __restrict__ Wp, const float* __restrict__ bih,
                                 const float* __restrict__ bhh, const float* __restrict__ bp,
                                 const float* __restrict__ Wih, bf16* __restrict__ Wpb,
                                 float* __restrict__ be, float* __restrict__ bd,
                                 unsigned* __restrict__ sync) {
  int e = blockIdx.x * blockDim.x + threadIdx.x;
  if (e < CD * CH) Wpb[e] = (bf16)Wp[e];
  if (e < G4) {
    float b0 = bih[e] + bhh[e];
    be[e] = b0;
    float acc = b0;
    #pragma unroll
    for (int j = 0; j < CD; ++j) acc += Wih[(size_t)e * CD + j] * bp[j];
    bd[e] = acc;
  }
  if (e == 0) { sync[0] = 0u; sync[1] = 0u; }
}

// ================= persistent LSTM kernel =================
__global__ __launch_bounds__(128, 1)
void lstm_kernel(const bf16* __restrict__ xn, const bf16* __restrict__ We,
                 const bf16* __restrict__ Wd, const bf16* __restrict__ Wpb,
                 const float* __restrict__ be_g, const float* __restrict__ bd_g,
                 const float* __restrict__ bp, const float* __restrict__ mean,
                 const float* __restrict__ stdev, const float* __restrict__ rw,
                 const float* __restrict__ rb, bf16* __restrict__ hbuf,
                 float* __restrict__ out, unsigned* __restrict__ sync) {
  const int tid  = threadIdx.x;
  const int lane = tid & 31;
  const int wid  = tid >> 5;          // 4 waves
  const int ji   = blockIdx.x & 15;   // unit strip
  const int bi   = blockIdx.x >> 4;   // batch group
  const int U0   = ji * UT;
  const int B0   = bi * BT;

  __shared__ float gsh[BT * 132];     // 32 x 128 gates, padded stride

  unsigned* cnt = sync;
  unsigned* gen = sync + 1;

  // zero my slice of h buffer 0 (initial hidden state)
  #pragma unroll
  for (int p = 0; p < 8; ++p) {
    int q = tid + 128 * p;
    hbuf[(size_t)(B0 + (q >> 5)) * CH + U0 + (q & 31)] = (bf16)0.0f;
  }

  // cell-update ownership: fixed u_local per thread, 8 batch rows
  const int ul = tid & 31;
  float be0 = be_g[0 * CH + U0 + ul], be1 = be_g[1 * CH + U0 + ul];
  float be2 = be_g[2 * CH + U0 + ul], be3 = be_g[3 * CH + U0 + ul];
  float bd0 = bd_g[0 * CH + U0 + ul], bd1 = bd_g[1 * CH + U0 + ul];
  float bd2 = bd_g[2 * CH + U0 + ul], bd3 = bd_g[3 * CH + U0 + ul];
  float cst[8];
  #pragma unroll
  for (int p = 0; p < 8; ++p) cst[p] = 0.0f;

  // per-lane weight-row base pointers for this wave's two N-tiles
  const bf16* wrE[2]; const bf16* wrD[2];
  #pragma unroll
  for (int i = 0; i < 2; ++i) {
    int ccol = (2 * wid + i) * 16 + (lane & 15);          // local gate col 0..127
    int grow = (ccol >> 5) * CH + U0 + (ccol & 31);       // global gate row
    wrE[i] = We + (size_t)grow * KCE;
    wrD[i] = Wd + (size_t)grow * CH;
  }
  const bf16* wproj_row = Wpb + (size_t)(lane & 15) * CH; // proj: N = feature d

  const v8f z8 = {0.f, 0.f, 0.f, 0.f, 0.f, 0.f, 0.f, 0.f};

  grid_sync(cnt, gen);

  int pp = 0;
  for (int step = 0; step < TIN + TOUT; ++step) {
    const bool enc = step < TIN;
    const bf16* hr = hbuf + (size_t)pp * CB * CH;
    bf16*       hw = hbuf + (size_t)(pp ^ 1) * CB * CH;
    const bf16* xrow = enc ? (xn + (size_t)step * CB * CD) : nullptr;

    // ---- decoder output y = denorm(proj(h)) : off the critical path ----
    if (!enc && ji == 0 && wid < 2) {
      int t = step - TIN;
      v8f pacc = z8;
      for (int kk = 0; kk < CH / 32; ++kk) {
        v16bf a = load_a(hr, nullptr, B0, wid, kk * 32, lane);
        v16bf b = load_b16(wproj_row, kk * 32, lane);
        pacc = wmma_bf16(a, b, pacc);
      }
      int d = lane & 15;
      float bpd = bp[d], rwd = rw[d], rbd = rb[d];
      #pragma unroll
      for (int r = 0; r < 8; ++r) {
        int gb = B0 + wid * 16 + r + ((lane & 16) ? 8 : 0);
        float pr = pacc[r] + bpd;
        float yv = (pr - rbd) / (rwd + 1e-10f) * stdev[gb * CD + d] + mean[gb * CD + d];
        out[(size_t)gb * CD * TOUT + (size_t)d * TOUT + t] = yv;
      }
    }

    // ---- recurrent GEMM: gates = [h | x_t] @ Wc^T ----
    v8f acc[2][2];
    acc[0][0] = z8; acc[0][1] = z8; acc[1][0] = z8; acc[1][1] = z8;
    const int nk = enc ? (KCE / 32) : (CH / 32);   // 17 or 16 k-tiles
    for (int kk = 0; kk < nk; ++kk) {
      int kb = kk * 32;
      v16bf a0 = load_a(hr, xrow, B0, 0, kb, lane);
      v16bf a1 = load_a(hr, xrow, B0, 1, kb, lane);
      v16bf b0 = load_b16(enc ? wrE[0] : wrD[0], kb, lane);
      v16bf b1 = load_b16(enc ? wrE[1] : wrD[1], kb, lane);
      acc[0][0] = wmma_bf16(a0, b0, acc[0][0]);
      acc[0][1] = wmma_bf16(a0, b1, acc[0][1]);
      acc[1][0] = wmma_bf16(a1, b0, acc[1][0]);
      acc[1][1] = wmma_bf16(a1, b1, acc[1][1]);
    }

    // ---- exchange gates through LDS ----
    __syncthreads();
    #pragma unroll
    for (int i = 0; i < 2; ++i) {
      int ntl = 2 * wid + i;
      #pragma unroll
      for (int m = 0; m < 2; ++m) {
        #pragma unroll
        for (int r = 0; r < 8; ++r) {
          int ml = m * 16 + r + ((lane & 16) ? 8 : 0);
          gsh[ml * 132 + ntl * 16 + (lane & 15)] = acc[m][i][r];
        }
      }
    }
    __syncthreads();

    // ---- LSTM cell update (c stays in registers for all 712 steps) ----
    float g0 = enc ? be0 : bd0, g1 = enc ? be1 : bd1;
    float g2 = enc ? be2 : bd2, g3 = enc ? be3 : bd3;
    #pragma unroll
    for (int p = 0; p < 8; ++p) {
      int bl = (tid >> 5) + 4 * p;
      const float* row = gsh + bl * 132;
      float gi = row[ul]          + g0;
      float gf = row[32 + ul]     + g1;
      float gg = row[64 + ul]     + g2;
      float go = row[96 + ul]     + g3;
      float c2 = sigmoidf_(gf) * cst[p] + sigmoidf_(gi) * tanhf(gg);
      cst[p] = c2;
      float h2 = sigmoidf_(go) * tanhf(c2);
      hw[(size_t)(B0 + bl) * CH + U0 + ul] = (bf16)h2;
    }

    grid_sync(cnt, gen);
    pp ^= 1;
  }
}

// ================= launcher =================
extern "C" void kernel_launch(void* const* d_in, const int* in_sizes, int n_in,
                              void* d_out, int out_size, void* d_ws, size_t ws_size,
                              hipStream_t stream) {
  const float* x      = (const float*)d_in[0];
  const float* W_ih   = (const float*)d_in[1];
  const float* W_hh   = (const float*)d_in[2];
  const float* b_ih   = (const float*)d_in[3];
  const float* b_hh   = (const float*)d_in[4];
  const float* W_proj = (const float*)d_in[5];
  const float* b_proj = (const float*)d_in[6];
  const float* rev_w  = (const float*)d_in[7];
  const float* rev_b  = (const float*)d_in[8];

  char* ws = (char*)d_ws;
  float*    mean  = (float*)(ws + OFF_MEAN);
  float*    stdev = (float*)(ws + OFF_STD);
  bf16*     xn    = (bf16*)(ws + OFF_XN);
  bf16*     We    = (bf16*)(ws + OFF_WE);
  bf16*     Wd    = (bf16*)(ws + OFF_WD);
  bf16*     Wpb   = (bf16*)(ws + OFF_WP);
  float*    be    = (float*)(ws + OFF_BE);
  float*    bd    = (float*)(ws + OFF_BD);
  bf16*     hbuf  = (bf16*)(ws + OFF_H);
  unsigned* sync  = (unsigned*)(ws + OFF_SYNC);
  float*    out   = (float*)d_out;

  stats_kernel<<<4096 / 4, 128, 0, stream>>>(x, mean, stdev);
  xn_kernel<<<(TIN * CB * CD + 255) / 256, 256, 0, stream>>>(x, mean, stdev, rev_w, rev_b, xn);
  pack_enc_kernel<<<(G4 * KCE + 255) / 256, 256, 0, stream>>>(W_hh, W_ih, We);
  pack_dec_kernel<<<(G4 * CH + 255) / 256, 256, 0, stream>>>(W_hh, W_ih, W_proj, Wd);
  pack_misc_kernel<<<(CD * CH + 255) / 256, 256, 0, stream>>>(W_proj, b_ih, b_hh, b_proj,
                                                              W_ih, Wpb, be, bd, sync);
  lstm_kernel<<<NWG, 128, 0, stream>>>(xn, We, Wd, Wpb, be, bd, b_proj, mean, stdev,
                                       rev_w, rev_b, hbuf, out, sync);
}
